// RNNmodel_21517786153005
// MI455X (gfx1250) — matrix operand order repaired and checked by reading, not compiled, not run on hardware
//
#include <hip/hip_runtime.h>
#include <hip/hip_bf16.h>
#include <math.h>

// Problem dims (match reference)
#define BB 128
#define TT 512
#define UU 512
#define GG 2048   // 4*U
#define OUTD 4

typedef __attribute__((ext_vector_type(16))) _Float16 v16h;
typedef __attribute__((ext_vector_type(8)))  _Float16 v8h;
typedef __attribute__((ext_vector_type(8)))  float    v8f;

union Frag16 { v16h v; v8h h[2]; };

// A operand (16x32 f16, M x K): lane l<16 holds row l, K = k0+0..7 and k0+16..23;
// lane l>=16 holds row (l-16), K = k0+8..15 and k0+24..31. Row-major source.
__device__ __forceinline__ v16h a_frag(const _Float16* rowbase, int k0, int hsel) {
    Frag16 f;
    f.h[0] = *(const v8h*)(rowbase + k0 + 8 * hsel);
    f.h[1] = *(const v8h*)(rowbase + k0 + 16 + 8 * hsel);
    return f.v;
}
// B operand (32x16 f16, K x N): lane l<16 holds col l, K = k0+0..7 and k0+16..23;
// lane l>=16 holds col (l-16), K = k0+8..15 and k0+24..31. Source is W transposed: WT[n][k].
__device__ __forceinline__ v16h b_frag(const _Float16* colbase, int k0, int hsel) {
    Frag16 f;
    const _Float16* p = colbase + k0 + 8 * hsel;
    f.h[0] = *(const v8h*)(p);
    f.h[1] = *(const v8h*)(p + 16);
    return f.v;
}

__device__ __forceinline__ float sigm(float x) { return 1.0f / (1.0f + expf(-x)); }

// One LSTM timestep: z = x_t @ WxT^T + h_prev @ WhT^T + b, gates, state update.
// Grid: (UU/16, BB/16), block: 32 threads (one wave). Each wave computes one
// 16(batch) x 16(unit) tile for ALL FOUR gates, then updates c/h in-register.
__global__ __launch_bounds__(32) void lstm_step_kernel(
    const _Float16* __restrict__ xseq,   // [B, T, Kx] row-major fp16
    const _Float16* __restrict__ WxT,    // [4U][Kx] fp16 (transposed)
    const _Float16* __restrict__ WhT,    // [4U][U]  fp16 (transposed)
    const float*    __restrict__ bias,   // [4U]
    const _Float16* __restrict__ h_prev, // [B][U] fp16
    float*          __restrict__ c_state,// [B][U] f32
    _Float16*       __restrict__ h_next, // [B][U] fp16
    _Float16*       __restrict__ hseq,   // optional output sequence (may be null)
    int t_in, int t_out, int Kx, int ldseq, int col_off, int Tn)
{
    const int lane = threadIdx.x;
    const int lh   = lane & 15;
    const int hsel = lane >> 4;
    const int u0   = blockIdx.x * 16;
    const int b0   = blockIdx.y * 16;

    // Bias broadcast: C-layout lane l holds column N = l&15 for all 8 M values.
    const float bi = bias[0 * UU + u0 + lh];
    const float bf = bias[1 * UU + u0 + lh];
    const float bg = bias[2 * UU + u0 + lh];
    const float bo = bias[3 * UU + u0 + lh];
    v8f acc_i = {bi, bi, bi, bi, bi, bi, bi, bi};
    v8f acc_f = {bf, bf, bf, bf, bf, bf, bf, bf};
    v8f acc_g = {bg, bg, bg, bg, bg, bg, bg, bg};
    v8f acc_o = {bo, bo, bo, bo, bo, bo, bo, bo};

    const _Float16* xrow = xseq + ((size_t)(b0 + lh) * Tn + t_in) * Kx;
    const _Float16* hrow = h_prev + (size_t)(b0 + lh) * UU;
    const _Float16* wxi = WxT + (size_t)(0 * UU + u0 + lh) * Kx;
    const _Float16* wxf = WxT + (size_t)(1 * UU + u0 + lh) * Kx;
    const _Float16* wxg = WxT + (size_t)(2 * UU + u0 + lh) * Kx;
    const _Float16* wxo = WxT + (size_t)(3 * UU + u0 + lh) * Kx;
    const _Float16* whi = WhT + (size_t)(0 * UU + u0 + lh) * UU;
    const _Float16* whf = WhT + (size_t)(1 * UU + u0 + lh) * UU;
    const _Float16* whg = WhT + (size_t)(2 * UU + u0 + lh) * UU;
    const _Float16* who = WhT + (size_t)(3 * UU + u0 + lh) * UU;

    // Input projection part: K over Kx
    for (int k0 = 0; k0 < Kx; k0 += 32) {
        v16h a  = a_frag(xrow, k0, hsel);
        v16h bI = b_frag(wxi, k0, hsel);
        v16h bF = b_frag(wxf, k0, hsel);
        v16h bG = b_frag(wxg, k0, hsel);
        v16h bO = b_frag(wxo, k0, hsel);
        acc_i = __builtin_amdgcn_wmma_f32_16x16x32_f16(false, a, false, bI, (short)0, acc_i, false, false);
        acc_f = __builtin_amdgcn_wmma_f32_16x16x32_f16(false, a, false, bF, (short)0, acc_f, false, false);
        acc_g = __builtin_amdgcn_wmma_f32_16x16x32_f16(false, a, false, bG, (short)0, acc_g, false, false);
        acc_o = __builtin_amdgcn_wmma_f32_16x16x32_f16(false, a, false, bO, (short)0, acc_o, false, false);
    }
    // Recurrent part: K over U
    for (int k0 = 0; k0 < UU; k0 += 32) {
        v16h a  = a_frag(hrow, k0, hsel);
        v16h bI = b_frag(whi, k0, hsel);
        v16h bF = b_frag(whf, k0, hsel);
        v16h bG = b_frag(whg, k0, hsel);
        v16h bO = b_frag(who, k0, hsel);
        acc_i = __builtin_amdgcn_wmma_f32_16x16x32_f16(false, a, false, bI, (short)0, acc_i, false, false);
        acc_f = __builtin_amdgcn_wmma_f32_16x16x32_f16(false, a, false, bF, (short)0, acc_f, false, false);
        acc_g = __builtin_amdgcn_wmma_f32_16x16x32_f16(false, a, false, bG, (short)0, acc_g, false, false);
        acc_o = __builtin_amdgcn_wmma_f32_16x16x32_f16(false, a, false, bO, (short)0, acc_o, false, false);
    }

    // Epilogue: C/D layout element r -> M = r + 8*hsel, N = lh
    const int uc = u0 + lh;
    #pragma unroll
    for (int r = 0; r < 8; ++r) {
        const int b = b0 + r + 8 * hsel;
        const size_t idx = (size_t)b * UU + uc;
        const float iv = sigm(acc_i[r]);
        const float fv = sigm(acc_f[r]);
        const float gv = tanhf(acc_g[r]);
        const float ov = sigm(acc_o[r]);
        const float cn = fv * c_state[idx] + iv * gv;
        const float hv = ov * tanhf(cn);
        c_state[idx] = cn;
        h_next[idx]  = (_Float16)hv;
        if (hseq)
            hseq[((size_t)b * Tn + t_out) * ldseq + col_off + uc] = (_Float16)hv;
    }
}

// Pack weight [K x 2048] f32 -> transposed fp16 [2048 x Kpad], zero-padded in K.
__global__ void pack_wT_kernel(const float* __restrict__ W, _Float16* __restrict__ WT,
                               int K, int Kpad, int total) {
    int i = blockIdx.x * blockDim.x + threadIdx.x;
    if (i >= total) return;
    int k = i % Kpad;
    int n = i / Kpad;
    WT[(size_t)n * Kpad + k] = (k < K) ? (_Float16)W[(size_t)k * GG + n] : (_Float16)0.0f;
}

// x [B,T,4] f32 -> [B,T,32] fp16 zero-padded (so layer-1 K-loop is a clean 32).
__global__ void pack_x_kernel(const float* __restrict__ x, _Float16* __restrict__ x16, int total) {
    int i = blockIdx.x * blockDim.x + threadIdx.x;
    if (i >= total) return;
    int k = i & 31;
    int bt = i >> 5;
    x16[i] = (k < 4) ? (_Float16)x[(size_t)bt * 4 + k] : (_Float16)0.0f;
}

__global__ void zero_state_kernel(_Float16* h0, _Float16* h1, float* c, int n) {
    int i = blockIdx.x * blockDim.x + threadIdx.x;
    if (i >= n) return;
    h0[i] = (_Float16)0.0f;
    h1[i] = (_Float16)0.0f;
    c[i]  = 0.0f;
}

// Final dense head: [128,512] x [512,4] — tiny, one thread per output element.
__global__ void dense_kernel(const _Float16* __restrict__ h, const float* __restrict__ Wd,
                             const float* __restrict__ bd, float* __restrict__ out) {
    int i = blockIdx.x * blockDim.x + threadIdx.x;
    if (i >= BB * OUTD) return;
    int b = i >> 2, o = i & 3;
    float s = bd[o];
    for (int k = 0; k < UU; ++k)
        s += (float)h[(size_t)b * UU + k] * Wd[(size_t)k * OUTD + o];
    out[i] = s;
}

extern "C" void kernel_launch(void* const* d_in, const int* in_sizes, int n_in,
                              void* d_out, int out_size, void* d_ws, size_t ws_size,
                              hipStream_t stream) {
    const float* x    = (const float*)d_in[0];
    const float* Wx1  = (const float*)d_in[1];
    const float* Wh1  = (const float*)d_in[2];
    const float* b1   = (const float*)d_in[3];
    const float* Wx2  = (const float*)d_in[4];
    const float* Wh2  = (const float*)d_in[5];
    const float* b2   = (const float*)d_in[6];
    const float* Wxf3 = (const float*)d_in[7];
    const float* Whf3 = (const float*)d_in[8];
    const float* bf3  = (const float*)d_in[9];
    const float* Wxb3 = (const float*)d_in[10];
    const float* Whb3 = (const float*)d_in[11];
    const float* bb3  = (const float*)d_in[12];
    const float* Wx4  = (const float*)d_in[13];
    const float* Wh4  = (const float*)d_in[14];
    const float* b4   = (const float*)d_in[15];
    const float* Wd   = (const float*)d_in[16];
    const float* bd   = (const float*)d_in[17];

    // Workspace carve-out (all sizes 256B-aligned)
    char* ws = (char*)d_ws;
    size_t off = 0;
    auto carve = [&](size_t bytes) -> char* {
        char* p = ws + off;
        off += (bytes + 255) & ~(size_t)255;
        return p;
    };
    _Float16* x16   = (_Float16*)carve((size_t)BB * TT * 32 * 2);
    _Float16* WxT1  = (_Float16*)carve((size_t)GG * 32 * 2);
    _Float16* WhT1  = (_Float16*)carve((size_t)GG * UU * 2);
    _Float16* WxT2  = (_Float16*)carve((size_t)GG * UU * 2);
    _Float16* WhT2  = (_Float16*)carve((size_t)GG * UU * 2);
    _Float16* WxTf3 = (_Float16*)carve((size_t)GG * UU * 2);
    _Float16* WhTf3 = (_Float16*)carve((size_t)GG * UU * 2);
    _Float16* WxTb3 = (_Float16*)carve((size_t)GG * UU * 2);
    _Float16* WhTb3 = (_Float16*)carve((size_t)GG * UU * 2);
    _Float16* WxT4  = (_Float16*)carve((size_t)GG * 2 * UU * 2);
    _Float16* WhT4  = (_Float16*)carve((size_t)GG * UU * 2);
    _Float16* h1seq = (_Float16*)carve((size_t)BB * TT * UU * 2);
    _Float16* h2seq = (_Float16*)carve((size_t)BB * TT * UU * 2);
    _Float16* h3seq = (_Float16*)carve((size_t)BB * TT * 2 * UU * 2);
    _Float16* hp0   = (_Float16*)carve((size_t)BB * UU * 2);
    _Float16* hp1   = (_Float16*)carve((size_t)BB * UU * 2);
    float*    cst   = (float*)carve((size_t)BB * UU * 4);
    (void)ws_size; (void)in_sizes; (void)n_in; (void)out_size;

    const int TPB = 256;
    // Pack x and all weights to fp16 (transposed) once per call
    {
        int tot = BB * TT * 32;
        pack_x_kernel<<<(tot + TPB - 1) / TPB, TPB, 0, stream>>>(x, x16, tot);
    }
    auto packW = [&](const float* W, _Float16* WT, int K, int Kpad) {
        int tot = GG * Kpad;
        pack_wT_kernel<<<(tot + TPB - 1) / TPB, TPB, 0, stream>>>(W, WT, K, Kpad, tot);
    };
    packW(Wx1, WxT1, 4, 32);
    packW(Wh1, WhT1, UU, UU);
    packW(Wx2, WxT2, UU, UU);
    packW(Wh2, WhT2, UU, UU);
    packW(Wxf3, WxTf3, UU, UU);
    packW(Whf3, WhTf3, UU, UU);
    packW(Wxb3, WxTb3, UU, UU);
    packW(Whb3, WhTb3, UU, UU);
    packW(Wx4, WxT4, 2 * UU, 2 * UU);
    packW(Wh4, WhT4, UU, UU);

    // One full LSTM pass over T timesteps (sequential chain, ping-pong h)
    auto run_layer = [&](const _Float16* xseq, int Kx, const _Float16* WxT,
                         const _Float16* WhT, const float* bias, _Float16* hseq,
                         int ldseq, int col_off, bool rev) -> const _Float16* {
        int n = BB * UU;
        zero_state_kernel<<<(n + TPB - 1) / TPB, TPB, 0, stream>>>(hp0, hp1, cst, n);
        _Float16* hp = hp0;
        _Float16* hn = hp1;
        for (int s = 0; s < TT; ++s) {
            int t = rev ? (TT - 1 - s) : s;
            lstm_step_kernel<<<dim3(UU / 16, BB / 16), 32, 0, stream>>>(
                xseq, WxT, WhT, bias, hp, cst, hn, hseq, t, t, Kx, ldseq, col_off, TT);
            _Float16* tmp = hp; hp = hn; hn = tmp;
        }
        return hp; // h_T (T even => ends in hp0)
    };

    run_layer(x16,   32,     WxT1,  WhT1,  b1,  h1seq, UU,     0,  false); // layer 1
    run_layer(h1seq, UU,     WxT2,  WhT2,  b2,  h2seq, UU,     0,  false); // layer 2
    run_layer(h2seq, UU,     WxTf3, WhTf3, bf3, h3seq, 2 * UU, 0,  false); // layer 3 fwd
    run_layer(h2seq, UU,     WxTb3, WhTb3, bb3, h3seq, 2 * UU, UU, true);  // layer 3 bwd
    const _Float16* h4 =
        run_layer(h3seq, 2 * UU, WxT4, WhT4, b4, nullptr, 1, 0, false);    // layer 4

    dense_kernel<<<(BB * OUTD + TPB - 1) / TPB, TPB, 0, stream>>>(h4, Wd, bd, (float*)d_out);
}
